// SwinFiTBlock_87222195847486
// MI455X (gfx1250) — matrix operand order repaired and checked by reading, not compiled, not run on hardware
//
#include <hip/hip_runtime.h>
#include <hip/hip_bf16.h>

typedef _Float16 f16;
typedef __attribute__((ext_vector_type(16))) _Float16 v16h;
typedef __attribute__((ext_vector_type(8)))  _Float16 v8h;
typedef __attribute__((ext_vector_type(8)))  float    v8f;

union F16x { v16h v; v8h h8[2]; f16 h[16]; };
union H8   { v8h v; f16 h[8]; };
union F8   { v8f v; float f[8]; };

#define C_DIM     768
#define BATCH     8
#define L_DIM     4096
#define HEADS     12
#define HD        64
#define SHIFT_    4
#define MROWS     32768   /* B * L = total window-tokens */
#define QKV_N     2304
#define HID_DIM   3072
#define MODS      4608

// ---------------------------------------------------------------------------
// WMMA fragment loaders (layouts per CDNA5 ISA 7.12.2, wave32)
// A (16x32 f16, row-major src, stride in halves):
//   lane<16: M=lane,    K = {0..7, 16..23}
//   lane>=16: M=lane-16, K = {8..15, 24..31}
// B (32x16 f16) loaded from an N-major ("Bt[n][k]") buffer:
//   lane holds N=lane&15, K = {0..7, 16..23} (lane<16) / {8..15, 24..31} (lane>=16)
// ---------------------------------------------------------------------------
__device__ __forceinline__ v16h load_frag_a(const f16* base, int stride, int k0, int lane) {
    int m  = lane & 15;
    int kb = (lane & 16) ? 8 : 0;
    const f16* p = base + m * stride + k0 + kb;
    F16x r;
    r.h8[0] = *(const v8h*)(p);
    r.h8[1] = *(const v8h*)(p + 16);
    return r.v;
}
__device__ __forceinline__ v16h load_frag_b(const f16* base, int stride, int k0, int lane) {
    int n  = lane & 15;
    int kb = (lane & 16) ? 8 : 0;
    const f16* p = base + n * stride + k0 + kb;
    F16x r;
    r.h8[0] = *(const v8h*)(p);
    r.h8[1] = *(const v8h*)(p + 16);
    return r.v;
}

// window-token row -> (batch, flat L index) with shift-roll undone
__device__ __forceinline__ void row_to_bl(int row, int& b, int& l) {
    int wi = row >> 6, t = row & 63;
    b = wi >> 6;
    int widx = wi & 63;
    int hr = ((widx >> 3) << 3) + (t >> 3);
    int wr = ((widx & 7) << 3) + (t & 7);
    int h = (hr + SHIFT_) & 63;
    int w = (wr + SHIFT_) & 63;
    l = h * 64 + w;
}

// ---------------------------------------------------------------------------
// adaLN modulation: mod[b, :] = silu(c[b]) @ w_adaln + b_adaln
// ---------------------------------------------------------------------------
__global__ void adaln_kernel(const float* __restrict__ c, const float* __restrict__ w,
                             const float* __restrict__ bias, float* __restrict__ mod) {
    __shared__ float sc[C_DIM];
    int b = blockIdx.x;
    for (int i = threadIdx.x; i < C_DIM; i += blockDim.x) {
        float v = c[b * C_DIM + i];
        sc[i] = v / (1.0f + __expf(-v));
    }
    __syncthreads();
    for (int j = threadIdx.x; j < MODS; j += blockDim.x) {
        float acc = bias[j];
        for (int i = 0; i < C_DIM; ++i) acc += sc[i] * w[i * MODS + j];
        mod[b * MODS + j] = acc;
    }
}

// ---------------------------------------------------------------------------
// LayerNorm + modulate (+ optional shifted-window gather) -> f16 activation
// ---------------------------------------------------------------------------
template <bool WIN>
__global__ void ln_mod_kernel(const float* __restrict__ src, const float* __restrict__ mod,
                              f16* __restrict__ dst) {
    __shared__ float red[256];
    __shared__ float stats[2];
    int row = blockIdx.x;
    int b, l;
    if (WIN) row_to_bl(row, b, l);
    else { b = row >> 12; l = row & 4095; }
    const float* xr = src + ((size_t)b * L_DIM + l) * C_DIM;

    float s1 = 0.f, s2 = 0.f;
    for (int i = threadIdx.x; i < C_DIM; i += 256) { float v = xr[i]; s1 += v; s2 += v * v; }
    red[threadIdx.x] = s1; __syncthreads();
    for (int st = 128; st > 0; st >>= 1) { if (threadIdx.x < st) red[threadIdx.x] += red[threadIdx.x + st]; __syncthreads(); }
    if (threadIdx.x == 0) stats[0] = red[0] * (1.0f / C_DIM);
    __syncthreads();
    red[threadIdx.x] = s2; __syncthreads();
    for (int st = 128; st > 0; st >>= 1) { if (threadIdx.x < st) red[threadIdx.x] += red[threadIdx.x + st]; __syncthreads(); }
    if (threadIdx.x == 0) {
        float mu = stats[0];
        float var = red[0] * (1.0f / C_DIM) - mu * mu;
        stats[1] = rsqrtf(var + 1e-6f);
    }
    __syncthreads();
    float mu = stats[0], rs = stats[1];
    const float* mb = mod + b * MODS + (WIN ? 0 : 2304);   // [shift | scale]
    f16* drow = dst + (size_t)row * C_DIM;
    for (int i = threadIdx.x; i < C_DIM; i += 256) {
        float v = (xr[i] - mu) * rs;
        v = v * (1.0f + mb[C_DIM + i]) + mb[i];
        drow[i] = (f16)v;
    }
}

// ---------------------------------------------------------------------------
__global__ void cvt_f16_kernel(const float* __restrict__ src, f16* __restrict__ dst, int n) {
    int i = blockIdx.x * 256 + threadIdx.x;
    if (i < n) dst[i] = (f16)src[i];
}

// ---------------------------------------------------------------------------
// Tiled f16 WMMA GEMM: Out(M,N) = A(M,K) @ Bw(K,N) + bias, fused epilogues.
// 256 threads = 8 waves; 64(M) x 128(N) block tile; wave = 16x64 strip
// (4 accumulators, one A-frag reused across 4 B-frags per K-step).
// MODE 0: store f16            (qkv)
// MODE 1: GELU, store f16      (fc1)
// MODE 2: proj fused: xout[b,l,c] = xin[b,l,c] + g_msa * v   (un-roll+residual)
// MODE 3: fc2 fused:  xout    += g_mlp * v   (in-place on x1/d_out)
// ---------------------------------------------------------------------------
template <int MODE>
__global__ void gemm_f16_kernel(const f16* __restrict__ A, const f16* __restrict__ Bw,
                                const float* __restrict__ bias,
                                f16* __restrict__ out16,
                                const float* __restrict__ xin, const float* __restrict__ mod,
                                float* __restrict__ xout,
                                int M, int N, int K) {
    __shared__ f16 As[64 * 32];    // [m][k]
    __shared__ f16 Bs[128 * 32];   // transposed: [n][k]
    const int tile_n = blockIdx.x * 128;
    const int tile_m = blockIdx.y * 64;
    const int wave = threadIdx.x >> 5, lane = threadIdx.x & 31;
    const int wm = (wave >> 1) * 16;      // 0,16,32,48
    const int wn = (wave & 1) * 64;       // 0,64

    F8 acc[4];
    #pragma unroll
    for (int t = 0; t < 4; ++t)
        for (int r = 0; r < 8; ++r) acc[t].f[r] = 0.f;

    const int ar = threadIdx.x >> 2, ac = (threadIdx.x & 3) * 8;   // A chunk: 64 rows x 4 x v8h

    for (int k0 = 0; k0 < K; k0 += 32) {
        // stage A tile (row-major, 16B vector loads)
        *(v8h*)&As[ar * 32 + ac] = *(const v8h*)&A[(size_t)(tile_m + ar) * K + k0 + ac];
        if (k0 + 32 < K)
            __builtin_prefetch(&A[(size_t)(tile_m + ar) * K + k0 + 32 + ac], 0, 1);
        // stage B tile transposed: 512 chunks of v8h (32 k x 16 n-groups), 2 per thread
        #pragma unroll
        for (int j = 0; j < 2; ++j) {
            const int cc = threadIdx.x + j * 256;
            const int kr = cc & 31, ng = cc >> 5;          // k row, n-group of 8
            H8 bv; bv.v = *(const v8h*)&Bw[(size_t)(k0 + kr) * N + tile_n + ng * 8];
            #pragma unroll
            for (int i = 0; i < 8; ++i) Bs[(ng * 8 + i) * 32 + kr] = bv.h[i];
        }
        __syncthreads();

        v16h af = load_frag_a(&As[wm * 32], 32, 0, lane);
        #pragma unroll
        for (int t = 0; t < 4; ++t) {
            v16h bf = load_frag_b(&Bs[(wn + t * 16) * 32], 32, 0, lane);
            acc[t].v = __builtin_amdgcn_wmma_f32_16x16x32_f16(false, af, false, bf, (short)0, acc[t].v, false, false);
        }
        __syncthreads();
    }

    // epilogue
    const int hi = (lane >> 4) * 8;
    const int nn = lane & 15;
    #pragma unroll
    for (int t = 0; t < 4; ++t) {
        const int col = tile_n + wn + t * 16 + nn;
        const float bcol = bias[col];
        #pragma unroll
        for (int r = 0; r < 8; ++r) {
            const int grow = tile_m + wm + hi + r;
            float v = acc[t].f[r] + bcol;
            if (MODE == 1) v = 0.5f * v * (1.0f + erff(v * 0.70710678118654752f));
            if (MODE == 0 || MODE == 1) {
                out16[(size_t)grow * N + col] = (f16)v;
            } else if (MODE == 2) {
                int b, l; row_to_bl(grow, b, l);
                size_t o = ((size_t)b * L_DIM + l) * C_DIM + col;
                float g = mod[b * MODS + 1536 + col];
                xout[o] = xin[o] + g * v;
            } else { // MODE 3
                int b = grow >> 12;
                size_t o = (size_t)grow * C_DIM + col;
                float g = mod[b * MODS + 3840 + col];
                xout[o] = xout[o] + g * v;
            }
        }
    }
}

// ---------------------------------------------------------------------------
// Windowed attention: one block per (window, head). 128 threads = 4 waves.
// S = (Q*0.125) Kt + rel_bias + shift_mask ; P = softmax(S) ; O = P V
// ---------------------------------------------------------------------------
__global__ void attention_kernel(const f16* __restrict__ qkv,
                                 const int* __restrict__ rel_index,
                                 const float* __restrict__ rpb,
                                 const float* __restrict__ maskm,
                                 f16* __restrict__ Ao) {
    __shared__ f16 Qs[64 * 64];
    __shared__ f16 Ks[64 * 64];
    __shared__ f16 Vt[64 * 64];         // transposed: Vt[d][n]
    __shared__ float Sm[4][16 * 64];
    __shared__ f16  Pm[4][16 * 64];

    const int wi = blockIdx.x;          // 0..511
    const int head = blockIdx.y;        // 0..11
    const int tid = threadIdx.x;
    const int wave = tid >> 5, lane = tid & 31;
    const int qoff = head * HD;

    // stage Q, K (row-major) and V (transposed) from qkv16 (row-major, stride 2304)
    for (int c = tid; c < 512; c += 128) {      // 64x64 halves in 8-wide chunks
        const int n = c >> 3, d8 = (c & 7) * 8;
        const size_t rb = (size_t)(wi * 64 + n) * QKV_N;
        *(v8h*)&Qs[n * 64 + d8] = *(const v8h*)&qkv[rb + qoff + d8];
        *(v8h*)&Ks[n * 64 + d8] = *(const v8h*)&qkv[rb + 768 + qoff + d8];
        H8 vv; vv.v = *(const v8h*)&qkv[rb + 1536 + qoff + d8];
        #pragma unroll
        for (int i = 0; i < 8; ++i) Vt[(d8 + i) * 64 + n] = vv.h[i];
    }
    __syncthreads();

    const int m0 = wave * 16;

    // S strip: 16x64
    F8 s[4];
    #pragma unroll
    for (int t = 0; t < 4; ++t)
        for (int r = 0; r < 8; ++r) s[t].f[r] = 0.f;
    #pragma unroll
    for (int d0 = 0; d0 < 64; d0 += 32) {
        v16h af = load_frag_a(&Qs[m0 * 64], 64, d0, lane);
        #pragma unroll
        for (int t = 0; t < 4; ++t) {
            v16h bf = load_frag_b(&Ks[t * 16 * 64], 64, d0, lane);
            s[t].v = __builtin_amdgcn_wmma_f32_16x16x32_f16(false, af, false, bf, (short)0, s[t].v, false, false);
        }
    }
    const int hi = (lane >> 4) * 8, nn = lane & 15;
    #pragma unroll
    for (int t = 0; t < 4; ++t)
        for (int r = 0; r < 8; ++r)
            Sm[wave][(hi + r) * 64 + t * 16 + nn] = s[t].f[r];
    __syncthreads();

    // scale + bias + mask + softmax (16 rows per wave, one lane per row)
    if (lane < 16) {
        const int mg = m0 + lane;                       // token in window
        float* srow = &Sm[wave][lane * 64];
        const int* ri = rel_index + mg * 64;
        const float* mrow = maskm + ((size_t)(wi & 63) * 64 + mg) * 64;
        float mx = -1e30f;
        for (int n = 0; n < 64; ++n) {
            float t = srow[n] * 0.125f + rpb[ri[n] * HEADS + head] + mrow[n];
            srow[n] = t;
            mx = fmaxf(mx, t);
        }
        float sum = 0.f;
        for (int n = 0; n < 64; ++n) { float e = __expf(srow[n] - mx); srow[n] = e; sum += e; }
        float inv = 1.0f / sum;
        f16* prow = &Pm[wave][lane * 64];
        for (int n = 0; n < 64; ++n) prow[n] = (f16)(srow[n] * inv);
    }
    __syncthreads();

    // O strip = P (16x64) @ V (64x64)
    F8 o[4];
    #pragma unroll
    for (int t = 0; t < 4; ++t)
        for (int r = 0; r < 8; ++r) o[t].f[r] = 0.f;
    #pragma unroll
    for (int n0 = 0; n0 < 64; n0 += 32) {
        v16h af = load_frag_a(&Pm[wave][0], 64, n0, lane);
        #pragma unroll
        for (int t = 0; t < 4; ++t) {
            v16h bf = load_frag_b(&Vt[t * 16 * 64], 64, n0, lane);
            o[t].v = __builtin_amdgcn_wmma_f32_16x16x32_f16(false, af, false, bf, (short)0, o[t].v, false, false);
        }
    }
    #pragma unroll
    for (int t = 0; t < 4; ++t)
        for (int r = 0; r < 8; ++r) {
            const int ml = hi + r;
            const int d = t * 16 + nn;
            Ao[(size_t)(wi * 64 + m0 + ml) * C_DIM + qoff + d] = (f16)o[t].f[r];
        }
}

// ---------------------------------------------------------------------------
extern "C" void kernel_launch(void* const* d_in, const int* in_sizes, int n_in,
                              void* d_out, int out_size, void* d_ws, size_t ws_size,
                              hipStream_t stream) {
    const float* x        = (const float*)d_in[0];
    const float* c        = (const float*)d_in[1];
    const float* maskm    = (const float*)d_in[2];
    const int*   rel_idx  = (const int*)d_in[3];
    const float* w_adaln  = (const float*)d_in[4];
    const float* b_adaln  = (const float*)d_in[5];
    const float* w_qkv    = (const float*)d_in[6];
    const float* b_qkv    = (const float*)d_in[7];
    const float* rpb      = (const float*)d_in[8];
    const float* w_proj   = (const float*)d_in[9];
    const float* b_proj   = (const float*)d_in[10];
    const float* w_fc1    = (const float*)d_in[11];
    const float* b_fc1    = (const float*)d_in[12];
    const float* w_fc2    = (const float*)d_in[13];
    const float* b_fc2    = (const float*)d_in[14];
    float* out = (float*)d_out;

    // workspace bump allocator (256B aligned)
    char* ws = (char*)d_ws;
    size_t off = 0;
    auto alloc = [&](size_t bytes) -> char* {
        char* p = ws + off;
        off = (off + bytes + 255) & ~(size_t)255;
        return p;
    };
    float* mod   = (float*)alloc((size_t)BATCH * MODS * 4);
    f16*   w16   = (f16*)  alloc((size_t)C_DIM * HID_DIM * 2);            // reused per GEMM
    f16*   A16   = (f16*)  alloc((size_t)MROWS * HID_DIM * 2);            // winh16, later fc1-out
    f16*   qkv16 = (f16*)  alloc((size_t)MROWS * QKV_N * 2);              // qkv
    f16*   B16   = (f16*)  alloc((size_t)MROWS * C_DIM * 2);              // attn-out, later h2

    // 1) adaLN modulation
    adaln_kernel<<<BATCH, 256, 0, stream>>>(c, w_adaln, b_adaln, mod);

    // 2) LN + modulate + roll + window partition -> A16 (f16)
    ln_mod_kernel<true><<<MROWS, 256, 0, stream>>>(x, mod, A16);

    // 3) QKV GEMM: (32768 x 768) @ (768 x 2304) -> qkv16
    cvt_f16_kernel<<<(C_DIM * QKV_N + 255) / 256, 256, 0, stream>>>(w_qkv, w16, C_DIM * QKV_N);
    gemm_f16_kernel<0><<<dim3(QKV_N / 128, MROWS / 64), 256, 0, stream>>>(
        A16, w16, b_qkv, qkv16, nullptr, nullptr, nullptr, MROWS, QKV_N, C_DIM);

    // 4) windowed attention -> B16 (f16)
    attention_kernel<<<dim3(512, HEADS), 128, 0, stream>>>(qkv16, rel_idx, rpb, maskm, B16);

    // 5) proj GEMM + fused un-roll/un-window + gated residual -> x1 in d_out
    cvt_f16_kernel<<<(C_DIM * C_DIM + 255) / 256, 256, 0, stream>>>(w_proj, w16, C_DIM * C_DIM);
    gemm_f16_kernel<2><<<dim3(C_DIM / 128, MROWS / 64), 256, 0, stream>>>(
        B16, w16, b_proj, nullptr, x, mod, out, MROWS, C_DIM, C_DIM);

    // 6) LN + modulate (MLP branch) on x1 -> B16 (f16)
    ln_mod_kernel<false><<<MROWS, 256, 0, stream>>>(out, mod, B16);

    // 7) fc1 GEMM + GELU -> A16 (f16)
    cvt_f16_kernel<<<(C_DIM * HID_DIM + 255) / 256, 256, 0, stream>>>(w_fc1, w16, C_DIM * HID_DIM);
    gemm_f16_kernel<1><<<dim3(HID_DIM / 128, MROWS / 64), 256, 0, stream>>>(
        B16, w16, b_fc1, A16, nullptr, nullptr, nullptr, MROWS, HID_DIM, C_DIM);

    // 8) fc2 GEMM + fused gated residual, in place on d_out (x1)
    cvt_f16_kernel<<<(HID_DIM * C_DIM + 255) / 256, 256, 0, stream>>>(w_fc2, w16, HID_DIM * C_DIM);
    gemm_f16_kernel<3><<<dim3(C_DIM / 128, MROWS / 64), 256, 0, stream>>>(
        A16, w16, b_fc2, nullptr, nullptr, mod, out, MROWS, C_DIM, HID_DIM);
}
